// Attention_82008105550161
// MI455X (gfx1250) — matrix operand order repaired and checked by reading, not compiled, not run on hardware
//
#include <hip/hip_runtime.h>

#define T_MEM   2
#define NTOK    64
#define CH      256
#define NH      8
#define HD      32

typedef __attribute__((ext_vector_type(16))) __bf16 v16bf;
typedef __attribute__((ext_vector_type(8)))  float  v8f;

union FragBF {
    v16bf  v;
    __bf16 e[16];
    uint4  q[2];
};

__device__ __forceinline__ v8f wmma_bf16(v16bf a, v16bf b, v8f c) {
    // D = A(16x32 bf16) x B(32x16 bf16) + C(f32)
    return __builtin_amdgcn_wmma_f32_16x16x32_bf16(false, a, false, b, (short)0, c, false, false);
}

__device__ __forceinline__ v8f vzero8() {
    v8f z = {0.f, 0.f, 0.f, 0.f, 0.f, 0.f, 0.f, 0.f};
    return z;
}

// A fragment (16x32 bf16) from row-major bf16 [row][stride].
// ISA: lane&15 = M row; lanes 16-31 use K offset +8; VGPR0-3 = K[koff..koff+7], VGPR4-7 = K[koff+16..koff+23]
__device__ __forceinline__ v16bf load_a_bf16(const __bf16* src, int row0, int stride, int k0, int lane) {
    int r    = row0 + (lane & 15);
    int koff = k0 + ((lane & 16) >> 1);
    const __bf16* p = src + r * stride + koff;
    FragBF f;
    f.q[0] = *(const uint4*)(p);
    f.q[1] = *(const uint4*)(p + 16);
    return f.v;
}

// B fragment (32x16): B[k][n], loaded from B^T stored row-major [n][stride] (i.e. weight W[n][k]).
// ISA: lane&15 = N col; lanes 16-31 use K offset +8; VGPR0-3 = K[koff..koff+7], VGPR4-7 = K[koff+16..koff+23]
__device__ __forceinline__ v16bf load_b_bf16(const __bf16* srcT, int n0, int stride, int k0, int lane) {
    int n    = n0 + (lane & 15);
    int koff = k0 + ((lane & 16) >> 1);
    const __bf16* p = srcT + n * stride + koff;
    FragBF f;
    f.q[0] = *(const uint4*)(p);
    f.q[1] = *(const uint4*)(p + 16);
    return f.v;
}

// A fragment from head-major O staging: base layout [h][64][HD]; global feature k -> (h = k/32, off = k%32).
// K-chunks (koff%32 in {0,8}; +16..+23 => 16..31) never cross a head boundary.
__device__ __forceinline__ v16bf load_a_headmajor(const __bf16* base, int row0, int k0, int lane) {
    int r    = row0 + (lane & 15);
    int koff = k0 + ((lane & 16) >> 1);
    int ka = koff, kb = koff + 16;
    const __bf16* pa = base + ((ka >> 5) << 11) + r * HD + (ka & 31);
    const __bf16* pb = base + ((kb >> 5) << 11) + r * HD + (kb & 31);
    FragBF f;
    f.q[0] = *(const uint4*)(pa);
    f.q[1] = *(const uint4*)(pb);
    return f.v;
}

// Cooperative fp32 [64][256] global tile -> bf16 [64][256] LDS tile (256 threads, coalesced)
__device__ __forceinline__ void stage_tile_bf16(const float* __restrict__ src,
                                                __bf16* __restrict__ dst, int tid) {
#pragma unroll
    for (int c = 0; c < 4; ++c) {
        int idx = c * 4096 + tid * 16;   // 4 * 256 * 16 = 16384 elements
        float tmp[16];
        *(float4*)(tmp + 0)  = *(const float4*)(src + idx + 0);
        *(float4*)(tmp + 4)  = *(const float4*)(src + idx + 4);
        *(float4*)(tmp + 8)  = *(const float4*)(src + idx + 8);
        *(float4*)(tmp + 12) = *(const float4*)(src + idx + 12);
        FragBF f;
#pragma unroll
        for (int i = 0; i < 16; ++i) f.e[i] = (__bf16)tmp[i];
        *(uint4*)(dst + idx + 0) = f.q[0];
        *(uint4*)(dst + idx + 8) = f.q[1];
    }
}

__global__ void convert_weights(const float* __restrict__ qw, const float* __restrict__ kvw,
                                const float* __restrict__ pw,
                                __bf16* __restrict__ wsq, __bf16* __restrict__ wskv,
                                __bf16* __restrict__ wsp) {
    int i = blockIdx.x * blockDim.x + threadIdx.x;   // 131072 threads
    if (i < CH * CH) wsq[i] = (__bf16)qw[i];
    if (i < 2 * CH * CH) wskv[i] = (__bf16)kvw[i];
    if (i < CH * CH) wsp[i] = (__bf16)pw[i];
}

__global__ __launch_bounds__(256, 1)
void attn_fused(const float* __restrict__ x, const float* __restrict__ memory,
                const __bf16* __restrict__ qw, const float* __restrict__ qb,
                const __bf16* __restrict__ kvw, const float* __restrict__ kvb,
                const __bf16* __restrict__ pw, const float* __restrict__ pb,
                const float* __restrict__ rpb, float* __restrict__ out)
{
    __shared__ __align__(16) __bf16 ks[NH][NTOK][HD];   // 32 KB: per-head K; reused as q / P staging
    __shared__ __align__(16) __bf16 vs[NH][NTOK * HD];  // 32 KB: staged input tile -> per-head V^T -> O staging

    const int b    = blockIdx.x;
    const int tid  = (int)threadIdx.x;
    const int w    = tid >> 5;            // wave id == head id
    const int lane = tid & 31;
    const int nl   = lane & 15;           // D-frag column within tile
    const int mofs = (lane >> 4) << 3;    // D-frag row offset (upper half -> +8)

    const float scale = 0.17677669529663687f;  // hd^-0.5

    __bf16* kw_lds = &ks[w][0][0];
    __bf16* vw_lds = &vs[w][0];
    __bf16* tile   = &vs[0][0];           // shared 64x256 bf16 staging view

    // ---------- Phase 0: cooperative stage of x ----------
    stage_tile_bf16(x + (size_t)b * (NTOK * CH), tile, tid);
    __syncthreads();

    // ---------- Phase 1: q_h = x @ q_w[h]^T + q_b[h]; keep as A-fragments in registers ----------
    v16bf aq[4];
    {
        v8f qa0[4], qa1[4];
#pragma unroll
        for (int mt = 0; mt < 4; ++mt) { qa0[mt] = vzero8(); qa1[mt] = vzero8(); }
#pragma unroll
        for (int k0 = 0; k0 < CH; k0 += 32) {
            v16bf b0 = load_b_bf16(qw, w * HD, CH, k0, lane);
            v16bf b1 = load_b_bf16(qw, w * HD + 16, CH, k0, lane);
#pragma unroll
            for (int mt = 0; mt < 4; ++mt) {
                v16bf a = load_a_bf16(tile, mt * 16, CH, k0, lane);
                qa0[mt] = wmma_bf16(a, b0, qa0[mt]);
                qa1[mt] = wmma_bf16(a, b1, qa1[mt]);
            }
        }
        float bq0 = qb[w * HD + nl];
        float bq1 = qb[w * HD + 16 + nl];
#pragma unroll
        for (int mt = 0; mt < 4; ++mt)
#pragma unroll
            for (int j = 0; j < 8; ++j) {   // D-frag -> [m][d] staging (wave-private region)
                int m = mt * 16 + mofs + j;
                kw_lds[m * HD + nl]      = (__bf16)(qa0[mt][j] + bq0);
                kw_lds[m * HD + 16 + nl] = (__bf16)(qa1[mt][j] + bq1);
            }
#pragma unroll
        for (int mt = 0; mt < 4; ++mt) aq[mt] = load_a_bf16(kw_lds, mt * 16, HD, 0, lane);
    }

    for (int t = 0; t < T_MEM; ++t) {
        const float* mb = memory + (size_t)(b * T_MEM + t) * (NTOK * CH);

        // ---------- Stage memory[b,t] into the shared tile ----------
        __syncthreads();                       // prior readers of vs are done (q-phase / proj-phase)
        stage_tile_bf16(mb, tile, tid);
        if (t + 1 < T_MEM) {                   // prefetch next memory tile into caches
            const float* nb = memory + (size_t)(b * T_MEM + t + 1) * (NTOK * CH);
            __builtin_prefetch(nb + tid * 32, 0, 1);
            __builtin_prefetch(nb + (256 + tid) * 32, 0, 1);
        }
        __syncthreads();

        // ---------- KV projection (head-w columns), results held in registers ----------
        v8f kvacc[4][4];                       // [n-tile: k0,k1,v0,v1][mt]
#pragma unroll
        for (int nt = 0; nt < 4; ++nt)
#pragma unroll
            for (int mt = 0; mt < 4; ++mt) kvacc[nt][mt] = vzero8();
#pragma unroll
        for (int k0 = 0; k0 < CH; k0 += 32) {
            v16bf bk0 = load_b_bf16(kvw, w * HD, CH, k0, lane);
            v16bf bk1 = load_b_bf16(kvw, w * HD + 16, CH, k0, lane);
            v16bf bv0 = load_b_bf16(kvw, CH + w * HD, CH, k0, lane);
            v16bf bv1 = load_b_bf16(kvw, CH + w * HD + 16, CH, k0, lane);
#pragma unroll
            for (int mt = 0; mt < 4; ++mt) {
                v16bf a = load_a_bf16(tile, mt * 16, CH, k0, lane);
                kvacc[0][mt] = wmma_bf16(a, bk0, kvacc[0][mt]);
                kvacc[1][mt] = wmma_bf16(a, bk1, kvacc[1][mt]);
                kvacc[2][mt] = wmma_bf16(a, bv0, kvacc[2][mt]);
                kvacc[3][mt] = wmma_bf16(a, bv1, kvacc[3][mt]);
            }
        }
        __syncthreads();                       // everyone done reading the staged tile

        // ---------- Write K -> ks[w][key][d], V^T -> vs[w][d][key] ----------
        {
            float bk0 = kvb[w * HD + nl];
            float bk1 = kvb[w * HD + 16 + nl];
            float bv0 = kvb[CH + w * HD + nl];
            float bv1 = kvb[CH + w * HD + 16 + nl];
#pragma unroll
            for (int mt = 0; mt < 4; ++mt)
#pragma unroll
                for (int j = 0; j < 8; ++j) {
                    int key = mt * 16 + mofs + j;
                    kw_lds[key * HD + nl]          = (__bf16)(kvacc[0][mt][j] + bk0);
                    kw_lds[key * HD + 16 + nl]     = (__bf16)(kvacc[1][mt][j] + bk1);
                    vw_lds[nl * NTOK + key]        = (__bf16)(kvacc[2][mt][j] + bv0);   // V^T[d][key]
                    vw_lds[(16 + nl) * NTOK + key] = (__bf16)(kvacc[3][mt][j] + bv1);
                }
        }

        // ---------- S = (q @ k^T)*scale + relative-position bias ----------
        v8f s[4][4];
#pragma unroll
        for (int nt = 0; nt < 4; ++nt) {
            v16bf bk = load_b_bf16(kw_lds, nt * 16, HD, 0, lane);  // B^T = k[key][d] row-major
#pragma unroll
            for (int mt = 0; mt < 4; ++mt) {
                v8f z = vzero8();
                s[mt][nt] = wmma_bf16(aq[mt], bk, z);
            }
        }
#pragma unroll
        for (int mt = 0; mt < 4; ++mt)
#pragma unroll
            for (int nt = 0; nt < 4; ++nt)
#pragma unroll
                for (int j = 0; j < 8; ++j) {
                    int m = mt * 16 + mofs + j;        // query token
                    int n = nt * 16 + nl;              // key token
                    int idx = ((m >> 3) - (n >> 3) + 7) * 15 + ((m & 7) - (n & 7) + 7);
                    s[mt][nt][j] = s[mt][nt][j] * scale + rpb[idx * NH + w];
                }

        // ---------- Softmax over 64 keys (per-lane partial over 4 nt + 16-lane shuffle reduce) ----------
#pragma unroll
        for (int mt = 0; mt < 4; ++mt)
#pragma unroll
            for (int j = 0; j < 8; ++j) {
                float mx = fmaxf(fmaxf(s[mt][0][j], s[mt][1][j]), fmaxf(s[mt][2][j], s[mt][3][j]));
#pragma unroll
                for (int d = 1; d < 16; d <<= 1) mx = fmaxf(mx, __shfl_xor(mx, d, 32));
                float sum = 0.f;
#pragma unroll
                for (int nt = 0; nt < 4; ++nt) {
                    float e = __expf(s[mt][nt][j] - mx);
                    s[mt][nt][j] = e;
                    sum += e;
                }
#pragma unroll
                for (int d = 1; d < 16; d <<= 1) sum += __shfl_xor(sum, d, 32);
                float inv = 1.0f / sum;
#pragma unroll
                for (int nt = 0; nt < 4; ++nt) s[mt][nt][j] *= inv;
            }

        // ---------- O = P @ V : stage P 32-key halves through LDS (k region is dead now) ----------
        v8f o[4][2];
#pragma unroll
        for (int mt = 0; mt < 4; ++mt) { o[mt][0] = vzero8(); o[mt][1] = vzero8(); }
#pragma unroll
        for (int kh = 0; kh < 2; ++kh) {
#pragma unroll
            for (int mt = 0; mt < 4; ++mt)
#pragma unroll
                for (int q2 = 0; q2 < 2; ++q2)
#pragma unroll
                    for (int j = 0; j < 8; ++j) {
                        int m = mt * 16 + mofs + j;
                        kw_lds[m * HD + q2 * 16 + nl] = (__bf16)s[mt][2 * kh + q2][j];
                    }
            v16bf bv0 = load_b_bf16(vw_lds, 0, NTOK, kh * 32, lane);   // B^T = V^T[d][key]
            v16bf bv1 = load_b_bf16(vw_lds, 16, NTOK, kh * 32, lane);
#pragma unroll
            for (int mt = 0; mt < 4; ++mt) {
                v16bf ap = load_a_bf16(kw_lds, mt * 16, HD, 0, lane);
                o[mt][0] = wmma_bf16(ap, bv0, o[mt][0]);
                o[mt][1] = wmma_bf16(ap, bv1, o[mt][1]);
            }
        }

        // ---------- Stage O_h into vs[w] as [m][d] (V^T fully consumed) ----------
#pragma unroll
        for (int mt = 0; mt < 4; ++mt)
#pragma unroll
            for (int nvt = 0; nvt < 2; ++nvt)
#pragma unroll
                for (int j = 0; j < 8; ++j) {
                    int m = mt * 16 + mofs + j;
                    vw_lds[m * HD + nvt * 16 + nl] = (__bf16)o[mt][nvt][j];
                }
        __syncthreads();

        // ---------- Output projection: wave w produces columns [32w, 32w+32) ----------
        float* ob = out + (size_t)(b * T_MEM + t) * (NTOK * CH);
        const __bf16* osall = &vs[0][0];
        v8f p0[4], p1[4];
#pragma unroll
        for (int mt = 0; mt < 4; ++mt) { p0[mt] = vzero8(); p1[mt] = vzero8(); }
#pragma unroll
        for (int k0 = 0; k0 < CH; k0 += 32) {
            v16bf b0 = load_b_bf16(pw, w * HD, CH, k0, lane);
            v16bf b1 = load_b_bf16(pw, w * HD + 16, CH, k0, lane);
#pragma unroll
            for (int mt = 0; mt < 4; ++mt) {
                v16bf a = load_a_headmajor(osall, mt * 16, k0, lane);
                p0[mt] = wmma_bf16(a, b0, p0[mt]);
                p1[mt] = wmma_bf16(a, b1, p1[mt]);
            }
        }
        {
            float pb0 = pb[w * HD + nl];
            float pb1 = pb[w * HD + 16 + nl];
#pragma unroll
            for (int mt = 0; mt < 4; ++mt)
#pragma unroll
                for (int j = 0; j < 8; ++j) {
                    int m = mt * 16 + mofs + j;
                    ob[(size_t)m * CH + w * HD + nl]      = p0[mt][j] + pb0;
                    ob[(size_t)m * CH + w * HD + 16 + nl] = p1[mt][j] + pb1;
                }
        }
    }
}

extern "C" void kernel_launch(void* const* d_in, const int* in_sizes, int n_in,
                              void* d_out, int out_size, void* d_ws, size_t ws_size,
                              hipStream_t stream) {
    (void)in_sizes; (void)n_in; (void)out_size; (void)ws_size;
    const float* x      = (const float*)d_in[0];
    const float* memory = (const float*)d_in[1];
    const float* q_w    = (const float*)d_in[2];
    const float* q_b    = (const float*)d_in[3];
    const float* kv_w   = (const float*)d_in[4];
    const float* kv_b   = (const float*)d_in[5];
    const float* proj_w = (const float*)d_in[6];
    const float* proj_b = (const float*)d_in[7];
    const float* rpb    = (const float*)d_in[8];

    __bf16* wsq  = (__bf16*)d_ws;          // 256*256 bf16
    __bf16* wskv = wsq + CH * CH;          // 512*256 bf16
    __bf16* wsp  = wskv + 2 * CH * CH;     // 256*256 bf16

    convert_weights<<<512, 256, 0, stream>>>(q_w, kv_w, proj_w, wsq, wskv, wsp);
    attn_fused<<<2048, 256, 0, stream>>>(x, memory, wsq, q_b, wskv, kv_b, wsp, proj_b,
                                         rpb, (float*)d_out);
}
